// LongformerPairClassifier_86586540688064
// MI455X (gfx1250) — matrix operand order repaired and checked by reading, not compiled, not run on hardware
//
#include <hip/hip_runtime.h>
#include <math.h>

// ---------------- constants (match reference) ----------------
#define BATCH  2
#define SEQ    2048
#define HID    768
#define NHEAD  12
#define HD     64
#define NGLOB  16
#define NLAYER 4
#define NPAIR  128
#define FFN    3072

typedef _Float16 v16h __attribute__((ext_vector_type(16)));
typedef _Float16 v8h  __attribute__((ext_vector_type(8)));
typedef float    v8f  __attribute__((ext_vector_type(8)));
typedef unsigned int v4u __attribute__((ext_vector_type(4)));
typedef int          v4i __attribute__((ext_vector_type(4)));
typedef int          v8i __attribute__((ext_vector_type(8)));

// D = A(16x32 f16) * B(32x16 f16) + C(16x16 f32)
#define WMMA(a, b, c) \
  __builtin_amdgcn_wmma_f32_16x16x32_f16(false, (a), false, (b), (short)0, (c), false, false)

// Tensor Data Mover path (gfx1250): guarded so either toolchain or a builtin-less
// compiler still builds via the vectorized-copy fallback.
#if defined(__HIP_DEVICE_COMPILE__) && __has_builtin(__builtin_amdgcn_tensor_load_to_lds)
#define USE_TDM 1
#else
#define USE_TDM 0
#endif

// A-fragment (16x32 f16): lanes 0-15 (row=lane) hold K 0..7 / 16..23,
// lanes 16-31 hold K 8..15 / 24..31. `stride` in halves; 16B-aligned chunks.
__device__ __forceinline__ v16h frag_a16(const _Float16* t, int stride, int lane) {
  const _Float16* p = t + (size_t)(lane & 15) * stride + ((lane & 16) ? 8 : 0);
  union { v16h v; v8h h[2]; } u;
  u.h[0] = *(const v8h*)(p);
  u.h[1] = *(const v8h*)(p + 16);
  return u.v;
}
// B-fragment (32x16 f16): lanes 0-15 (col=lane) hold K 0..7 / 16..23, lanes 16-31 K 8..15 / 24..31.
__device__ __forceinline__ v16h frag_b16(const _Float16* t, int stride, int lane) {
  const _Float16* p = t + (size_t)(lane & 15) * stride + ((lane & 16) ? 8 : 0);
  union { v16h v; v8h h[2]; } u;
  u.h[0] = *(const v8h*)(p);
  u.h[1] = *(const v8h*)(p + 16);
  return u.v;
}

#if USE_TDM
// Issue one TDM 2-D tile load: f16 tensor, row-major [rows][K], tile tileR x tileK,
// landing contiguously in LDS (== As[tileR][tileK] layout). D# per ISA ch.8.
__device__ __forceinline__ void tdm_load_2d(const _Float16* g, unsigned lds_byte,
                                            int K, int tileK, int tileR) {
  unsigned long long ga = (unsigned long long)(size_t)g;
  v4u g0;
  g0[0] = 1u;                                   // count=1, user descriptor
  g0[1] = lds_byte;                             // lds_addr (bytes)
  g0[2] = (unsigned)(ga & 0xffffffffu);         // global_addr[31:0]
  g0[3] = (unsigned)((ga >> 32) & 0x01ffffffu)  // global_addr[56:32]
          | (2u << 30);                         // type = 2 ("image")
  unsigned K_u = (unsigned)K, R_u = (unsigned)tileR;
  v8i g1;
  g1[0] = (int)(1u << 16);                      // data_size=1 (2 bytes); no flags
  g1[1] = (int)((K_u & 0xffffu) << 16);         // tensor_dim0[15:0] @ bits 63:48
  g1[2] = (int)(((K_u >> 16) & 0xffffu) | ((R_u & 0xffffu) << 16)); // dim0 hi / dim1 lo
  g1[3] = (int)(((R_u >> 16) & 0xffffu) | ((unsigned)tileK << 16)); // dim1 hi / tile_dim0
  g1[4] = (int)(R_u & 0xffffu);                 // tile_dim1 (tile_dim2 = 0)
  g1[5] = (int)K_u;                             // tensor_dim0_stride[31:0] (elements)
  g1[6] = 0;                                    // stride hi / dim1_stride lo
  g1[7] = 0;
  v4i z4 = {0, 0, 0, 0};
#if __clang_major__ >= 23
  v8i z8 = {0, 0, 0, 0, 0, 0, 0, 0};
  __builtin_amdgcn_tensor_load_to_lds(g0, g1, z4, z4, z8, 0);
#else
  __builtin_amdgcn_tensor_load_to_lds(g0, g1, z4, z4, 0);
#endif
}
#endif

// ---------------- embedding gather (f32 + f16 shadow) ----------------
__global__ __launch_bounds__(256) void embed_kernel(
    const int* __restrict__ ids, const float* __restrict__ emb,
    float* __restrict__ x, _Float16* __restrict__ xh) {
  const int row = blockIdx.x, tid = threadIdx.x;
  const int id = ids[row];
  for (int c = tid; c < HID; c += 256) {
    float v = emb[(size_t)id * HID + c];
    x[(size_t)row * HID + c] = v;
    xh[(size_t)row * HID + c] = (_Float16)v;
  }
}

// ---------------- weight transpose + f16 convert: Wt[n][k] = (f16)W[k][n] -----
__global__ __launch_bounds__(256) void transpose_cvt_kernel(
    const float* __restrict__ W, _Float16* __restrict__ Wt, int K, int N) {
  __shared__ float tile[32][33];
  const int nb = blockIdx.x * 32, kb = blockIdx.y * 32;
  const int tx = threadIdx.x & 31, ty = threadIdx.x >> 5;  // ty in 0..7
  for (int i = ty; i < 32; i += 8)
    tile[i][tx] = W[(size_t)(kb + i) * N + nb + tx];
  __syncthreads();
  for (int i = ty; i < 32; i += 8)
    Wt[(size_t)(nb + i) * K + kb + tx] = (_Float16)tile[tx][i];
}

// ---------------- WMMA GEMM with TDM-fed, double-buffered LDS ----------------
// out[M,N] = act(A[M,K]_f16 @ Wt[N,K]_f16^T + bias) * scale ; out f32 or f16.
__device__ __forceinline__ void stage_tiles_copy(
    _Float16 (*As)[32], _Float16 (*Bs)[32],
    const _Float16* __restrict__ A, const _Float16* __restrict__ Wt,
    int bm, int bn, int K, int k0, int tid) {
  const int r = tid >> 1, ch = (tid & 1) * 16;
  *(v8h*)&As[r][ch]     = *(const v8h*)(A  + (size_t)(bm + r) * K + k0 + ch);
  *(v8h*)&As[r][ch + 8] = *(const v8h*)(A  + (size_t)(bm + r) * K + k0 + ch + 8);
  *(v8h*)&Bs[r][ch]     = *(const v8h*)(Wt + (size_t)(bn + r) * K + k0 + ch);
  *(v8h*)&Bs[r][ch + 8] = *(const v8h*)(Wt + (size_t)(bn + r) * K + k0 + ch + 8);
}

__global__ __launch_bounds__(256) void gemm_kernel(
    const _Float16* __restrict__ A, const _Float16* __restrict__ Wt,
    const float* __restrict__ bias, void* __restrict__ out,
    int N, int K, float scale, int act, int out16) {
  __shared__ _Float16 As[2][128][32];
  __shared__ _Float16 Bs[2][128][32];
  const int tid = threadIdx.x, lane = tid & 31, wid = tid >> 5;
  const int bm = blockIdx.y * 128, bn = blockIdx.x * 128;
  const int wm = (wid & 3) * 32, wn = (wid >> 2) * 64;
  const int nk = K >> 5;
  v8f acc[2][4] = {};

#if USE_TDM
  const unsigned ldsA[2] = {(unsigned)(size_t)&As[0][0][0], (unsigned)(size_t)&As[1][0][0]};
  const unsigned ldsB[2] = {(unsigned)(size_t)&Bs[0][0][0], (unsigned)(size_t)&Bs[1][0][0]};
  if (wid == 0) {
    tdm_load_2d(A  + (size_t)bm * K, ldsA[0], K, 32, 128);
    tdm_load_2d(Wt + (size_t)bn * K, ldsB[0], K, 32, 128);
  }
  for (int kt = 0; kt < nk; ++kt) {
    const int buf = kt & 1;
    if (wid == 0) __builtin_amdgcn_s_wait_tensorcnt((short)0);
    __syncthreads();
    if (wid == 0 && kt + 1 < nk) {  // prefetch next tile via TDM while computing
      const int k0n = (kt + 1) << 5;
      tdm_load_2d(A  + (size_t)bm * K + k0n, ldsA[buf ^ 1], K, 32, 128);
      tdm_load_2d(Wt + (size_t)bn * K + k0n, ldsB[buf ^ 1], K, 32, 128);
    }
    v16h a0 = frag_a16(&As[buf][wm][0], 32, lane);
    v16h a1 = frag_a16(&As[buf][wm + 16][0], 32, lane);
    #pragma unroll
    for (int j = 0; j < 4; ++j) {
      v16h bfr = frag_b16(&Bs[buf][wn + j * 16][0], 32, lane);
      acc[0][j] = WMMA(a0, bfr, acc[0][j]);
      acc[1][j] = WMMA(a1, bfr, acc[1][j]);
    }
  }
#else
  stage_tiles_copy(As[0], Bs[0], A, Wt, bm, bn, K, 0, tid);
  for (int kt = 0; kt < nk; ++kt) {
    const int buf = kt & 1;
    __syncthreads();
    if (kt + 1 < nk)
      stage_tiles_copy(As[buf ^ 1], Bs[buf ^ 1], A, Wt, bm, bn, K, (kt + 1) << 5, tid);
    v16h a0 = frag_a16(&As[buf][wm][0], 32, lane);
    v16h a1 = frag_a16(&As[buf][wm + 16][0], 32, lane);
    #pragma unroll
    for (int j = 0; j < 4; ++j) {
      v16h bfr = frag_b16(&Bs[buf][wn + j * 16][0], 32, lane);
      acc[0][j] = WMMA(a0, bfr, acc[0][j]);
      acc[1][j] = WMMA(a1, bfr, acc[1][j]);
    }
  }
#endif

  #pragma unroll
  for (int mi = 0; mi < 2; ++mi)
    #pragma unroll
    for (int j = 0; j < 4; ++j)
      #pragma unroll
      for (int e = 0; e < 8; ++e) {
        int row = bm + wm + mi * 16 + ((lane >> 4) << 3) + e;
        int col = bn + wn + j * 16 + (lane & 15);
        float val = acc[mi][j][e] + bias[col];
        if (act) {  // tanh-approx GELU (jax.nn.gelu default)
          float x3 = val * val * val;
          val = 0.5f * val * (1.0f + tanhf(0.7978845608028654f * (val + 0.044715f * x3)));
        }
        val *= scale;
        if (out16) ((_Float16*)out)[(size_t)row * N + col] = (_Float16)val;
        else       ((float*)out)[(size_t)row * N + col] = val;
      }
}

// ---------------- band + global-key attention (WMMA, f16 I/O) ----------------
__global__ __launch_bounds__(64) void band_attn_kernel(
    const _Float16* __restrict__ q, const _Float16* __restrict__ k,
    const _Float16* __restrict__ v, _Float16* __restrict__ o) {
  __shared__ _Float16 Qs[32][64];
  __shared__ _Float16 Ks[32][64];
  __shared__ float    Ss[32][400];
  __shared__ _Float16 Vt[64][40];
  const int h = blockIdx.x, qblk = blockIdx.y, b = blockIdx.z;
  const int n = qblk >> 2;
  const int q0 = (qblk & 3) * 32;
  const int s0 = n * 128 + q0;
  const int tid = threadIdx.x, lane = tid & 31, wid = tid >> 5;

  for (int i = 0; i < 32; ++i) {
    int Lx = i * 64 + tid; int r = Lx >> 6, c = Lx & 63;
    Qs[r][c] = q[((size_t)(b * SEQ + s0 + r)) * HID + h * HD + c];
  }
  __syncthreads();
  v16h a0 = frag_a16(&Qs[wid * 16][0], 64, lane);
  v16h a1 = frag_a16(&Qs[wid * 16][32], 64, lane);

  for (int c = 0; c < 13; ++c) {
    int kvb = c * 32;
    for (int i = 0; i < 32; ++i) {
      int Lx = i * 64 + tid; int r = Lx >> 6, cc = Lx & 63;
      int kv = kvb + r;
      _Float16 val = (_Float16)0.0f;
      if (kv < 384) {
        int kpos = n * 128 + kv - 128;
        if (kpos >= 0 && kpos < SEQ)
          val = k[((size_t)(b * SEQ + kpos)) * HID + h * HD + cc];
      } else if (kv < 400) {
        val = k[((size_t)(b * SEQ + (kv - 384))) * HID + h * HD + cc];
      }
      Ks[r][cc] = val;
    }
    __syncthreads();
    for (int t = 0; t < 2; ++t) {
      v16h b0 = frag_b16(&Ks[t * 16][0], 64, lane);
      v16h b1 = frag_b16(&Ks[t * 16][32], 64, lane);
      v8f sacc = {};
      sacc = WMMA(a0, b0, sacc);
      sacc = WMMA(a1, b1, sacc);
      int col = kvb + t * 16 + (lane & 15);
      if (col < 400) {
        #pragma unroll
        for (int e = 0; e < 8; ++e) {
          int r = wid * 16 + ((lane >> 4) << 3) + e;
          bool ok;
          if (col < 384) {
            int kpos = n * 128 + col - 128;
            int rel = col - 128 - (q0 + r);
            ok = (rel >= -128) && (rel <= 128) && (kpos >= NGLOB) && (kpos < SEQ);
          } else {
            ok = true;
          }
          Ss[r][col] = ok ? sacc[e] : -1.0e9f;
        }
      }
    }
    __syncthreads();
  }

  if (tid < 32) {
    volatile float* srow = &Ss[tid][0];
    float m = -3.4e38f;
    for (int jj = 0; jj < 400; ++jj) { float sv = srow[jj]; m = fmaxf(m, sv); }
    float ssum = 0.0f;
    for (int jj = 0; jj < 400; ++jj) ssum += __expf(srow[jj] - m);
    float inv = 1.0f / ssum;
    volatile _Float16* pr = (volatile _Float16*)&Ss[tid][0];
    for (int jj = 0; jj < 400; ++jj) {
      float pv = __expf(srow[jj] - m) * inv;
      pr[jj] = (_Float16)pv;  // in-place f16 probs; byte 2j written after byte 4j read
    }
    for (int jj = 400; jj < 416; ++jj) pr[jj] = (_Float16)0.0f;
  }
  __syncthreads();

  v8f oacc[4] = {};
  const _Float16* Ph = (const _Float16*)&Ss[0][0];  // probs, row stride 800 halves
  for (int c = 0; c < 13; ++c) {
    int kvb = c * 32;
    for (int i = 0; i < 32; ++i) {
      int Lx = i * 64 + tid; int d = Lx >> 5, kvl = Lx & 31;
      int kv = kvb + kvl;
      _Float16 val = (_Float16)0.0f;
      if (kv < 384) {
        int kpos = n * 128 + kv - 128;
        if (kpos >= 0 && kpos < SEQ)
          val = v[((size_t)(b * SEQ + kpos)) * HID + h * HD + d];
      } else if (kv < 400) {
        val = v[((size_t)(b * SEQ + (kv - 384))) * HID + h * HD + d];
      }
      Vt[d][kvl] = val;
    }
    __syncthreads();
    v16h pa = frag_a16(Ph + (size_t)(wid * 16) * 800 + kvb, 800, lane);
    #pragma unroll
    for (int j = 0; j < 4; ++j) {
      v16h bv = frag_b16(&Vt[j * 16][0], 40, lane);
      oacc[j] = WMMA(pa, bv, oacc[j]);
    }
    __syncthreads();
  }
  #pragma unroll
  for (int j = 0; j < 4; ++j)
    #pragma unroll
    for (int e = 0; e < 8; ++e) {
      int r = wid * 16 + ((lane >> 4) << 3) + e;
      int d = j * 16 + (lane & 15);
      o[((size_t)(b * SEQ + s0 + r)) * HID + h * HD + d] = (_Float16)oacc[j][e];
    }
}

// ---------------- full attention for the G=16 global queries ----------------
__global__ __launch_bounds__(256) void glob_attn_kernel(
    const _Float16* __restrict__ q, const _Float16* __restrict__ k,
    const _Float16* __restrict__ v, _Float16* __restrict__ o) {
  const int g = blockIdx.x, h = blockIdx.y, b = blockIdx.z;
  __shared__ float sc[SEQ];
  __shared__ float red[256];
  __shared__ float qrow[HD];
  const int tid = threadIdx.x;
  const size_t qoff = ((size_t)(b * SEQ + g)) * HID + h * HD;
  if (tid < HD) qrow[tid] = (float)q[qoff + tid];
  __syncthreads();
  for (int s = tid; s < SEQ; s += 256) {
    size_t ko = ((size_t)(b * SEQ + s)) * HID + h * HD;
    float dot = 0.0f;
    #pragma unroll
    for (int d = 0; d < HD; ++d) dot += qrow[d] * (float)k[ko + d];
    sc[s] = dot;
  }
  __syncthreads();
  float m = -3.4e38f;
  for (int s = tid; s < SEQ; s += 256) m = fmaxf(m, sc[s]);
  red[tid] = m; __syncthreads();
  for (int off = 128; off > 0; off >>= 1) {
    if (tid < off) red[tid] = fmaxf(red[tid], red[tid + off]);
    __syncthreads();
  }
  m = red[0]; __syncthreads();
  float ps = 0.0f;
  for (int s = tid; s < SEQ; s += 256) { float e = __expf(sc[s] - m); sc[s] = e; ps += e; }
  red[tid] = ps; __syncthreads();
  for (int off = 128; off > 0; off >>= 1) {
    if (tid < off) red[tid] += red[tid + off];
    __syncthreads();
  }
  float inv = 1.0f / red[0];
  if (tid < HD) {
    float acc = 0.0f;
    for (int s = 0; s < SEQ; ++s)
      acc += sc[s] * (float)v[((size_t)(b * SEQ + s)) * HID + h * HD + tid];
    o[qoff + tid] = (_Float16)(acc * inv);
  }
}

// -------- residual + LayerNorm: x = LN(x + t)*g + b ; also writes f16 shadow --
__global__ __launch_bounds__(256) void ln_kernel(
    const float* __restrict__ x, const float* __restrict__ t,
    const float* __restrict__ g, const float* __restrict__ be,
    float* __restrict__ outp, _Float16* __restrict__ outh) {
  __shared__ float red[256];
  const int row = blockIdx.x, tid = threadIdx.x;
  float vals[3];
  #pragma unroll
  for (int i = 0; i < 3; ++i) {
    int c = tid + i * 256;
    vals[i] = x[(size_t)row * HID + c] + t[(size_t)row * HID + c];
  }
  red[tid] = vals[0] + vals[1] + vals[2]; __syncthreads();
  for (int off = 128; off > 0; off >>= 1) {
    if (tid < off) red[tid] += red[tid + off];
    __syncthreads();
  }
  float mean = red[0] / (float)HID; __syncthreads();
  float d0 = vals[0] - mean, d1 = vals[1] - mean, d2 = vals[2] - mean;
  red[tid] = d0 * d0 + d1 * d1 + d2 * d2; __syncthreads();
  for (int off = 128; off > 0; off >>= 1) {
    if (tid < off) red[tid] += red[tid + off];
    __syncthreads();
  }
  float inv = rsqrtf(red[0] / (float)HID + 1e-5f);
  #pragma unroll
  for (int i = 0; i < 3; ++i) {
    int c = tid + i * 256;
    float val = (vals[i] - mean) * inv * g[c] + be[c];
    outp[(size_t)row * HID + c] = val;
    outh[(size_t)row * HID + c] = (_Float16)val;
  }
}

// ---------------- pair head: concat(ei,ej) @ Wh + bh ----------------
__global__ __launch_bounds__(128) void head_kernel(
    const float* __restrict__ x, const int* __restrict__ pairs,
    const float* __restrict__ Wh, const float* __restrict__ bh,
    float* __restrict__ outp) {
  __shared__ float red[3][128];
  const int rp = blockIdx.x, tid = threadIdx.x;
  const int b = rp >> 7, p = rp & 127;
  const int i0 = pairs[(size_t)(b * NPAIR + p) * 2 + 0];
  const int i1 = pairs[(size_t)(b * NPAIR + p) * 2 + 1];
  float a0 = 0.0f, a1 = 0.0f, a2 = 0.0f;
  for (int j = tid; j < 2 * HID; j += 128) {
    float e = (j < HID) ? x[(size_t)(b * SEQ + i0) * HID + j]
                        : x[(size_t)(b * SEQ + i1) * HID + (j - HID)];
    a0 += e * Wh[(size_t)j * 3 + 0];
    a1 += e * Wh[(size_t)j * 3 + 1];
    a2 += e * Wh[(size_t)j * 3 + 2];
  }
  red[0][tid] = a0; red[1][tid] = a1; red[2][tid] = a2; __syncthreads();
  for (int off = 64; off > 0; off >>= 1) {
    if (tid < off) {
      red[0][tid] += red[0][tid + off];
      red[1][tid] += red[1][tid + off];
      red[2][tid] += red[2][tid + off];
    }
    __syncthreads();
  }
  if (tid < 3) outp[(size_t)rp * 3 + tid] = red[tid][0] + bh[tid];
}

// ---------------- orchestration ----------------
extern "C" void kernel_launch(void* const* d_in, const int* in_sizes, int n_in,
                              void* d_out, int out_size, void* d_ws, size_t ws_size,
                              hipStream_t stream) {
  (void)in_sizes; (void)n_in; (void)out_size; (void)ws_size;
  const int*   input_ids = (const int*)d_in[0];
  const int*   pairs     = (const int*)d_in[1];
  const float* emb       = (const float*)d_in[2];
  const float* Wq  = (const float*)d_in[3];  const float* bq  = (const float*)d_in[4];
  const float* Wk  = (const float*)d_in[5];  const float* bk  = (const float*)d_in[6];
  const float* Wv  = (const float*)d_in[7];  const float* bv  = (const float*)d_in[8];
  const float* Wo  = (const float*)d_in[9];  const float* bo  = (const float*)d_in[10];
  const float* g1  = (const float*)d_in[11]; const float* b1  = (const float*)d_in[12];
  const float* Wf1 = (const float*)d_in[13]; const float* bf1 = (const float*)d_in[14];
  const float* Wf2 = (const float*)d_in[15]; const float* bf2 = (const float*)d_in[16];
  const float* g2  = (const float*)d_in[17]; const float* b2  = (const float*)d_in[18];
  const float* Wh  = (const float*)d_in[19]; const float* bh  = (const float*)d_in[20];

  const size_t SLOT = (size_t)(BATCH * SEQ) * HID;  // 4096*768 elements
  float* ws0 = (float*)d_ws;
  float* x  = ws0;                 // f32 residual stream
  float* tb = ws0 + SLOT;          // f32 pre-LN branch
  _Float16* xh  = (_Float16*)(ws0 + 2 * SLOT);  // f16 shadow of x
  _Float16* qh  = xh + SLOT;
  _Float16* kh  = xh + 2 * SLOT;
  _Float16* vh  = xh + 3 * SLOT;
  _Float16* ah  = xh + 4 * SLOT;
  _Float16* hbh = qh;              // FFN hidden [4096][3072] aliases qh..ah (dead then)
  _Float16* wq_t  = xh + 5 * SLOT;                       // [768][768]
  _Float16* wk_t  = wq_t + (size_t)HID * HID;
  _Float16* wv_t  = wk_t + (size_t)HID * HID;
  _Float16* wo_t  = wv_t + (size_t)HID * HID;
  _Float16* wf1_t = wo_t + (size_t)HID * HID;            // [3072][768]
  _Float16* wf2_t = wf1_t + (size_t)HID * FFN;           // [768][3072]

  embed_kernel<<<dim3(BATCH * SEQ), 256, 0, stream>>>(input_ids, emb, x, xh);

  const float qscale = 0.125f;  // 1/sqrt(64)
  for (int l = 0; l < NLAYER; ++l) {
    const float* Wq_l  = Wq  + (size_t)l * HID * HID;  const float* bq_l  = bq  + (size_t)l * HID;
    const float* Wk_l  = Wk  + (size_t)l * HID * HID;  const float* bk_l  = bk  + (size_t)l * HID;
    const float* Wv_l  = Wv  + (size_t)l * HID * HID;  const float* bv_l  = bv  + (size_t)l * HID;
    const float* Wo_l  = Wo  + (size_t)l * HID * HID;  const float* bo_l  = bo  + (size_t)l * HID;
    const float* g1_l  = g1  + (size_t)l * HID;        const float* b1_l  = b1  + (size_t)l * HID;
    const float* Wf1_l = Wf1 + (size_t)l * HID * FFN;  const float* bf1_l = bf1 + (size_t)l * FFN;
    const float* Wf2_l = Wf2 + (size_t)l * FFN * HID;  const float* bf2_l = bf2 + (size_t)l * HID;
    const float* g2_l  = g2  + (size_t)l * HID;        const float* b2_l  = b2  + (size_t)l * HID;

    // transpose + f16-convert this layer's weights (Wt[n][k])
    transpose_cvt_kernel<<<dim3(HID / 32, HID / 32), 256, 0, stream>>>(Wq_l, wq_t, HID, HID);
    transpose_cvt_kernel<<<dim3(HID / 32, HID / 32), 256, 0, stream>>>(Wk_l, wk_t, HID, HID);
    transpose_cvt_kernel<<<dim3(HID / 32, HID / 32), 256, 0, stream>>>(Wv_l, wv_t, HID, HID);
    transpose_cvt_kernel<<<dim3(HID / 32, HID / 32), 256, 0, stream>>>(Wo_l, wo_t, HID, HID);
    transpose_cvt_kernel<<<dim3(FFN / 32, HID / 32), 256, 0, stream>>>(Wf1_l, wf1_t, HID, FFN);
    transpose_cvt_kernel<<<dim3(HID / 32, FFN / 32), 256, 0, stream>>>(Wf2_l, wf2_t, FFN, HID);

    gemm_kernel<<<dim3(HID / 128, 32), 256, 0, stream>>>(xh, wq_t, bq_l, qh, HID, HID, qscale, 0, 1);
    gemm_kernel<<<dim3(HID / 128, 32), 256, 0, stream>>>(xh, wk_t, bk_l, kh, HID, HID, 1.0f, 0, 1);
    gemm_kernel<<<dim3(HID / 128, 32), 256, 0, stream>>>(xh, wv_t, bv_l, vh, HID, HID, 1.0f, 0, 1);

    band_attn_kernel<<<dim3(NHEAD, SEQ / 32, BATCH), 64, 0, stream>>>(qh, kh, vh, ah);
    glob_attn_kernel<<<dim3(NGLOB, NHEAD, BATCH), 256, 0, stream>>>(qh, kh, vh, ah);

    gemm_kernel<<<dim3(HID / 128, 32), 256, 0, stream>>>(ah, wo_t, bo_l, tb, HID, HID, 1.0f, 0, 0);
    ln_kernel<<<dim3(BATCH * SEQ), 256, 0, stream>>>(x, tb, g1_l, b1_l, x, xh);

    gemm_kernel<<<dim3(FFN / 128, 32), 256, 0, stream>>>(xh, wf1_t, bf1_l, hbh, FFN, HID, 1.0f, 1, 1);
    gemm_kernel<<<dim3(HID / 128, 32), 256, 0, stream>>>(hbh, wf2_t, bf2_l, tb, HID, FFN, 1.0f, 0, 0);
    ln_kernel<<<dim3(BATCH * SEQ), 256, 0, stream>>>(x, tb, g2_l, b2_l, x, xh);
  }

  head_kernel<<<dim3(BATCH * NPAIR), 128, 0, stream>>>(x, pairs, Wh, bh, (float*)d_out);
}
